// ChimeraV2Block_78847009620427
// MI455X (gfx1250) — matrix operand order repaired and checked by reading, not compiled, not run on hardware
//
#include <hip/hip_runtime.h>

typedef _Float16 h16;
typedef __attribute__((ext_vector_type(16))) _Float16 v16h;
typedef __attribute__((ext_vector_type(8)))  float    v8f;

#define S_LEN   2048
#define D_MODEL 2048
#define H_Q     16
#define H_KV    4
#define HD      128
#define WIN     512

union FragH { v16h v; unsigned u[8]; h16 h[16]; };
union FragF { v8f  v; float f[8]; };

__device__ __forceinline__ v8f wmma16(v16h a, v16h b, v8f c) {
    return __builtin_amdgcn_wmma_f32_16x16x32_f16(false, a, false, b, (short)0, c, false, false);
}

// ---------------------------------------------------------------- convert
__global__ void k_f32_to_f16(const float* __restrict__ in, h16* __restrict__ out, int n) {
    int i = blockIdx.x * blockDim.x + threadIdx.x;
    if (i < n) out[i] = (h16)in[i];
}

// ---------------------------------------------------------------- GEMM (f16 x f16 -> OutT), C[M,N] = A[M,K] * B[K,N]
// 64x64 C tile per 128-thread block (4 waves). K-step 32. LDS stages A row-major
// and B transposed (n-major) so both fragment gathers are contiguous dword loads.
template <typename OutT>
__global__ __launch_bounds__(128) void k_gemm_f16(const h16* __restrict__ A,
                                                  const h16* __restrict__ B,
                                                  OutT* __restrict__ C,
                                                  int M, int N, int K) {
    __shared__ h16 As[64 * 40];   // 64 rows x 32 K (+8 pad)
    __shared__ h16 Bt[64 * 40];   // 64 n   x 32 K (+8 pad), transposed

    const int t      = threadIdx.x;
    const int lane   = t & 31;
    const int w      = t >> 5;
    const int laneHi = lane >> 4;
    const int lane15 = lane & 15;
    const int m0     = blockIdx.y * 64;
    const int n0     = blockIdx.x * 64;

    v8f acc[4];
    #pragma unroll
    for (int c = 0; c < 4; ++c) acc[c] = (v8f){0.f,0.f,0.f,0.f,0.f,0.f,0.f,0.f};

    for (int k0 = 0; k0 < K; k0 += 32) {
        __syncthreads();
        // stage A: 64x32 halves, 16B per thread-load, 2 loads/thread
        #pragma unroll
        for (int i = 0; i < 2; ++i) {
            int lin = t * 2 + i;
            int row = lin >> 2, seg = lin & 3;
            const float4* src = reinterpret_cast<const float4*>(A + (size_t)(m0 + row) * K + k0 + seg * 8);
            *reinterpret_cast<float4*>(&As[row * 40 + seg * 8]) = *src;
        }
        // stage B transposed: 32x64 -> Bt[n][k]
        unsigned short* bts = reinterpret_cast<unsigned short*>(Bt);
        #pragma unroll
        for (int i = 0; i < 8; ++i) {
            int p  = t * 8 + i;
            int kk = p >> 5;
            int n  = (p & 31) * 2;
            unsigned d = *reinterpret_cast<const unsigned*>(B + (size_t)(k0 + kk) * N + n0 + n);
            bts[n * 40 + kk]       = (unsigned short)(d & 0xFFFFu);
            bts[(n + 1) * 40 + kk] = (unsigned short)(d >> 16);
        }
        if (k0 + 32 < K) {  // gfx1250 global_prefetch_b8 of next tiles
            __builtin_prefetch(A + (size_t)(m0 + (t >> 1)) * K + k0 + 32, 0, 1);
            __builtin_prefetch(B + (size_t)(k0 + 32 + (t & 31)) * N + n0, 0, 1);
        }
        __syncthreads();

        // A fragment: 16x32, lanes 0-15/16-31 hold same rows, different K groups
        FragH a;
        const int arow = w * 16 + lane15;
        #pragma unroll
        for (int d = 0; d < 8; ++d) {
            int kk = 2 * d + 16 * laneHi;
            a.u[d] = *reinterpret_cast<const unsigned*>(&As[arow * 40 + kk]);
        }
        #pragma unroll
        for (int c = 0; c < 4; ++c) {
            FragH b;
            const unsigned* src = reinterpret_cast<const unsigned*>(&Bt[(c * 16 + lane15) * 40 + 16 * laneHi]);
            #pragma unroll
            for (int d = 0; d < 8; ++d) b.u[d] = src[d];
            acc[c] = wmma16(a.v, b.v, acc[c]);
        }
    }

    #pragma unroll
    for (int c = 0; c < 4; ++c) {
        FragF f; f.v = acc[c];
        #pragma unroll
        for (int r = 0; r < 8; ++r) {
            int row = m0 + w * 16 + r + 8 * laneHi;
            int col = n0 + c * 16 + lane15;
            C[(size_t)row * N + col] = (OutT)f.f[r];
        }
    }
}

// ---------------------------------------------------------------- RoPE:  lin[s][h*128+d] -> dst[h][s][128]
__global__ void k_rope(const h16* __restrict__ lin, h16* __restrict__ dst, int nh) {
    int i = blockIdx.x * blockDim.x + threadIdx.x;
    int total = nh * S_LEN * 64;
    if (i >= total) return;
    int d = i & 63;
    int s = (i >> 6) & (S_LEN - 1);
    int h = i >> 17;
    float inv = expf(-(float)d * 0.14391156831212787f);  // ln(10000)/64
    float ang = (float)s * inv;
    float sn, cs;
    sincosf(ang, &sn, &cs);
    size_t src = (size_t)s * (nh * HD) + h * HD + d;
    float x1 = (float)lin[src];
    float x2 = (float)lin[src + 64];
    size_t o = ((size_t)h * S_LEN + s) * HD + d;
    dst[o]      = (h16)(x1 * cs - x2 * sn);
    dst[o + 64] = (h16)(x2 * cs + x1 * sn);
}

// ---------------------------------------------------------------- V transpose: lin[s][kv*128+d] -> Vt[kv][d][s]
__global__ void k_vt(const h16* __restrict__ lin, h16* __restrict__ dst) {
    int i = blockIdx.x * blockDim.x + threadIdx.x;
    if (i >= H_KV * HD * S_LEN) return;
    int s  = i & (S_LEN - 1);
    int d  = (i >> 11) & (HD - 1);
    int kv = i >> 18;
    dst[i] = lin[(size_t)s * (H_KV * HD) + kv * HD + d];
}

// ---------------------------------------------------------------- windowed differential attention
// One wave per (head, 16-query tile). Two-pass online softmax on both
// head-dim halves, then p = relu(a1 - lam*a2), renorm, P@V via WMMA.
__global__ __launch_bounds__(32) void k_attn(const h16* __restrict__ Qr,   // [H][S][128]
                                             const h16* __restrict__ Kr,   // [Hk][S][128]
                                             const h16* __restrict__ Vt,   // [Hk][128][S]
                                             const float* __restrict__ lam,
                                             h16* __restrict__ out) {      // [S][H*128]
    __shared__ h16 pT[16 * 20];

    const int lane   = threadIdx.x & 31;
    const int laneHi = lane >> 4;
    const int lane15 = lane & 15;
    const int qb     = blockIdx.x * 16;
    const int h      = blockIdx.y;
    const int kv     = h >> 2;   // n_rep = 4
    const float lamh = lam[h];

    // Q fragments: 4 x (16x32) covering head_dim 0..127
    FragH qf[4];
    {
        const h16* qbase = Qr + ((size_t)h * S_LEN + qb + lane15) * HD;
        #pragma unroll
        for (int f = 0; f < 4; ++f)
            #pragma unroll
            for (int d = 0; d < 8; ++d) {
                int kk = f * 32 + 2 * d + 16 * laneHi;
                qf[f].u[d] = *reinterpret_cast<const unsigned*>(qbase + kk);
            }
    }

    int lo = qb - (WIN - 1); if (lo < 0) lo = 0;
    const int kt0 = lo >> 4;
    const int kt1 = qb >> 4;

    float m1[8], z1[8], m2[8], z2[8];
    #pragma unroll
    for (int r = 0; r < 8; ++r) { m1[r] = -1e30f; z1[r] = 0.f; m2[r] = -1e30f; z2[r] = 0.f; }

    // -------- pass 1: per-row max / sum-exp (per lane, over its column subset)
    for (int kt = kt0; kt <= kt1; ++kt) {
        FragH kf[4];
        const h16* kbase = Kr + ((size_t)kv * S_LEN + kt * 16 + lane15) * HD + 16 * laneHi;
        #pragma unroll
        for (int f = 0; f < 4; ++f) {
            const unsigned* src = reinterpret_cast<const unsigned*>(kbase + f * 32);
            #pragma unroll
            for (int d = 0; d < 8; ++d) kf[f].u[d] = src[d];
        }
        v8f s1 = (v8f){0.f,0.f,0.f,0.f,0.f,0.f,0.f,0.f};
        v8f s2 = (v8f){0.f,0.f,0.f,0.f,0.f,0.f,0.f,0.f};
        s1 = wmma16(qf[0].v, kf[0].v, s1);
        s1 = wmma16(qf[1].v, kf[1].v, s1);
        s2 = wmma16(qf[2].v, kf[2].v, s2);
        s2 = wmma16(qf[3].v, kf[3].v, s2);
        FragF f1, f2; f1.v = s1; f2.v = s2;
        #pragma unroll
        for (int r = 0; r < 8; ++r) {
            int row = qb + r + 8 * laneHi;
            int col = kt * 16 + lane15;
            if (col <= row && col >= row - (WIN - 1)) {
                float v1 = f1.f[r] * 0.125f, v2 = f2.f[r] * 0.125f;
                float mn = fmaxf(m1[r], v1);
                z1[r] = z1[r] * expf(m1[r] - mn) + expf(v1 - mn); m1[r] = mn;
                mn = fmaxf(m2[r], v2);
                z2[r] = z2[r] * expf(m2[r] - mn) + expf(v2 - mn); m2[r] = mn;
            }
        }
    }
    // merge stats across the 16-lane N group (butterfly)
    #pragma unroll
    for (int r = 0; r < 8; ++r) {
        #pragma unroll
        for (int msk = 1; msk < 16; msk <<= 1) {
            float mo = __shfl_xor(m1[r], msk, 32);
            float zo = __shfl_xor(z1[r], msk, 32);
            float mn = fmaxf(m1[r], mo);
            z1[r] = z1[r] * expf(m1[r] - mn) + zo * expf(mo - mn); m1[r] = mn;
            mo = __shfl_xor(m2[r], msk, 32);
            zo = __shfl_xor(z2[r], msk, 32);
            mn = fmaxf(m2[r], mo);
            z2[r] = z2[r] * expf(m2[r] - mn) + zo * expf(mo - mn); m2[r] = mn;
        }
        z1[r] = fmaxf(z1[r], 1e-20f);
        z2[r] = fmaxf(z2[r], 1e-20f);
    }

    // -------- pass 2: p = relu(a1 - lam*a2), accumulate den and P@V
    v8f oacc[8];
    #pragma unroll
    for (int c = 0; c < 8; ++c) oacc[c] = (v8f){0.f,0.f,0.f,0.f,0.f,0.f,0.f,0.f};
    float den[8];
    #pragma unroll
    for (int r = 0; r < 8; ++r) den[r] = 0.f;

    for (int kt = kt0; kt <= kt1; ++kt) {
        FragH kf[4];
        const h16* kbase = Kr + ((size_t)kv * S_LEN + kt * 16 + lane15) * HD + 16 * laneHi;
        #pragma unroll
        for (int f = 0; f < 4; ++f) {
            const unsigned* src = reinterpret_cast<const unsigned*>(kbase + f * 32);
            #pragma unroll
            for (int d = 0; d < 8; ++d) kf[f].u[d] = src[d];
        }
        v8f s1 = (v8f){0.f,0.f,0.f,0.f,0.f,0.f,0.f,0.f};
        v8f s2 = (v8f){0.f,0.f,0.f,0.f,0.f,0.f,0.f,0.f};
        s1 = wmma16(qf[0].v, kf[0].v, s1);
        s1 = wmma16(qf[1].v, kf[1].v, s1);
        s2 = wmma16(qf[2].v, kf[2].v, s2);
        s2 = wmma16(qf[3].v, kf[3].v, s2);
        FragF f1, f2; f1.v = s1; f2.v = s2;

        #pragma unroll
        for (int r = 0; r < 8; ++r) {
            int row = qb + r + 8 * laneHi;
            int col = kt * 16 + lane15;
            float p = 0.f;
            if (col <= row && col >= row - (WIN - 1)) {
                float a1 = expf(f1.f[r] * 0.125f - m1[r]) / z1[r];
                float a2 = expf(f2.f[r] * 0.125f - m2[r]) / z2[r];
                p = fmaxf(a1 - lamh * a2, 0.f);
            }
            den[r] += p;
            pT[(r + 8 * laneHi) * 20 + lane15] = (h16)p;   // C-layout -> row-major LDS
        }
        __syncthreads();

        // P as A fragment (16x32, keys 16..31 zero-padded)
        FragH pa;
        #pragma unroll
        for (int d = 0; d < 8; ++d) {
            int kk = 2 * d;
            pa.u[d] = laneHi ? 0u : *reinterpret_cast<const unsigned*>(&pT[lane15 * 20 + kk]);
        }

        #pragma unroll
        for (int c = 0; c < 8; ++c) {
            FragH vb;
            if (laneHi == 0) {   // rows k=0..15 = real keys; k=16..31 zero pad
                const unsigned* src = reinterpret_cast<const unsigned*>(
                    Vt + ((size_t)kv * HD + c * 16 + lane15) * S_LEN + kt * 16);
                #pragma unroll
                for (int d = 0; d < 8; ++d) vb.u[d] = src[d];
            } else {
                #pragma unroll
                for (int d = 0; d < 8; ++d) vb.u[d] = 0u;
            }
            oacc[c] = wmma16(pa.v, vb.v, oacc[c]);
        }
        __syncthreads();
    }

    #pragma unroll
    for (int r = 0; r < 8; ++r)
        #pragma unroll
        for (int msk = 1; msk < 16; msk <<= 1)
            den[r] += __shfl_xor(den[r], msk, 32);

    #pragma unroll
    for (int c = 0; c < 8; ++c) {
        FragF f; f.v = oacc[c];
        #pragma unroll
        for (int r = 0; r < 8; ++r) {
            int row = qb + r + 8 * laneHi;
            int col = c * 16 + lane15;
            out[(size_t)row * (H_Q * HD) + h * HD + col] = (h16)(f.f[r] / (den[r] + 1e-6f));
        }
    }
}

// ---------------------------------------------------------------- launch
extern "C" void kernel_launch(void* const* d_in, const int* in_sizes, int n_in,
                              void* d_out, int out_size, void* d_ws, size_t ws_size,
                              hipStream_t stream) {
    (void)in_sizes; (void)n_in; (void)out_size; (void)ws_size;
    const float* x   = (const float*)d_in[0];
    const float* Wq  = (const float*)d_in[1];
    const float* Wk  = (const float*)d_in[2];
    const float* Wv  = (const float*)d_in[3];
    const float* Wo  = (const float*)d_in[4];
    const float* lam = (const float*)d_in[5];
    float* out = (float*)d_out;

    char* ws = (char*)d_ws;
    const size_t MB = 1024ull * 1024ull;
    h16* xb   = (h16*)(ws + 0 * MB);    // 8 MB
    h16* Wqb  = (h16*)(ws + 8 * MB);    // 8 MB
    h16* Wkb  = (h16*)(ws + 16 * MB);   // 2 MB
    h16* Wvb  = (h16*)(ws + 18 * MB);   // 2 MB
    h16* Wob  = (h16*)(ws + 20 * MB);   // 8 MB
    h16* qlin = (h16*)(ws + 28 * MB);   // 8 MB  (reused as attn output)
    h16* klin = (h16*)(ws + 36 * MB);   // 2 MB
    h16* vlin = (h16*)(ws + 38 * MB);   // 2 MB
    h16* Qr   = (h16*)(ws + 40 * MB);   // 8 MB
    h16* Kr   = (h16*)(ws + 48 * MB);   // 2 MB
    h16* Vt   = (h16*)(ws + 50 * MB);   // 2 MB
    h16* attn = qlin;

    auto cvt = [&](const float* s, h16* d, int n) {
        k_f32_to_f16<<<(n + 255) / 256, 256, 0, stream>>>(s, d, n);
    };
    cvt(x,  xb,  D_MODEL * S_LEN);
    cvt(Wq, Wqb, D_MODEL * (H_Q * HD));
    cvt(Wk, Wkb, D_MODEL * (H_KV * HD));
    cvt(Wv, Wvb, D_MODEL * (H_KV * HD));
    cvt(Wo, Wob, (H_Q * HD) * D_MODEL);

    dim3 blk(128);
    k_gemm_f16<h16><<<dim3((H_Q * HD) / 64, S_LEN / 64), blk, 0, stream>>>(
        xb, Wqb, qlin, S_LEN, H_Q * HD, D_MODEL);
    k_gemm_f16<h16><<<dim3((H_KV * HD) / 64, S_LEN / 64), blk, 0, stream>>>(
        xb, Wkb, klin, S_LEN, H_KV * HD, D_MODEL);
    k_gemm_f16<h16><<<dim3((H_KV * HD) / 64, S_LEN / 64), blk, 0, stream>>>(
        xb, Wvb, vlin, S_LEN, H_KV * HD, D_MODEL);

    { int n = H_Q  * S_LEN * 64; k_rope<<<(n + 255) / 256, 256, 0, stream>>>(qlin, Qr, H_Q);  }
    { int n = H_KV * S_LEN * 64; k_rope<<<(n + 255) / 256, 256, 0, stream>>>(klin, Kr, H_KV); }
    { int n = H_KV * HD * S_LEN; k_vt<<<(n + 255) / 256, 256, 0, stream>>>(vlin, Vt); }

    k_attn<<<dim3(S_LEN / 16, H_Q), 32, 0, stream>>>(Qr, Kr, Vt, lam, attn);

    k_gemm_f16<float><<<dim3(D_MODEL / 64, S_LEN / 64), blk, 0, stream>>>(
        attn, Wob, out, S_LEN, D_MODEL, H_Q * HD);
}
